// Qwen3Attention_31439160606902
// MI455X (gfx1250) — matrix operand order repaired and checked by reading, not compiled, not run on hardware
//
#include <hip/hip_runtime.h>
#include <hip/hip_bf16.h>
#include <math.h>

typedef __attribute__((ext_vector_type(16))) _Float16 v16h;
typedef __attribute__((ext_vector_type(8)))  _Float16 v8h;
typedef __attribute__((ext_vector_type(4)))  _Float16 v4h;
typedef __attribute__((ext_vector_type(8)))  float    v8f;
typedef __attribute__((ext_vector_type(4)))  float    v4f;

#define B_    4
#define S_    16
#define HID_  4096
#define H_    32
#define HKV_  8
#define D_    128
#define PAST_ 4080
#define KLEN_ 4096
#define M_    64   // B_*S_ rows

__device__ __forceinline__ v8f wmma16x16x32(v16h a, v16h b, v8f c) {
  // (neg_a, A, neg_b, B, c_mod, C, reuse_a, reuse_b)
  return __builtin_amdgcn_wmma_f32_16x16x32_f16(false, a, false, b, (short)0, c,
                                                false, false);
}

__device__ __forceinline__ v16h cat16(v8h lo, v8h hi) {
  v16h r;
#pragma unroll
  for (int j = 0; j < 8; ++j) { r[j] = lo[j]; r[8 + j] = hi[j]; }
  return r;
}

// ---------------------------------------------------------------------------
// f32 -> f16 staging (vectorized 4-wide)
// ---------------------------------------------------------------------------
__global__ void f32_to_f16_kernel(const float* __restrict__ src,
                                  _Float16* __restrict__ dst, const int n4) {
  const int i = blockIdx.x * blockDim.x + threadIdx.x;
  if (i >= n4) return;
  v4f x = ((const v4f*)src)[i];
  v4h y;
#pragma unroll
  for (int j = 0; j < 4; ++j) y[j] = (_Float16)x[j];
  ((v4h*)dst)[i] = y;
}

// ---------------------------------------------------------------------------
// Generic M=64 GEMM: C[64,N] = Ah[64,K](f16) * W[K,N](f32) (+bias), f16 WMMA.
// One wave per 16-column tile; 4 M-tiles per wave => each W column read once.
// All loads indexed off fixed bases (immediate offsets; no flat pointers).
// ---------------------------------------------------------------------------
__global__ void gemm_f16wmma(const _Float16* __restrict__ Ah,
                             const float* __restrict__ W,
                             const float* __restrict__ bias,
                             float* __restrict__ C,
                             const int K, const int N) {
  const int n0   = blockIdx.x << 4;
  const int lane = threadIdx.x;
  const int half = lane >> 4;
  const int l15  = lane & 15;

  // constant per-lane element offsets
  const size_t a0 = (size_t)l15 * K + (half << 3);
  const size_t a1 = a0 + (size_t)16 * K;
  const size_t a2 = a0 + (size_t)32 * K;
  const size_t a3 = a0 + (size_t)48 * K;
  const float* wbase = W + (size_t)(half << 3) * N + n0 + l15;

  v8f acc0 = {}, acc1 = {}, acc2 = {}, acc3 = {};
  size_t wk = 0;                               // running k-offset into W
  for (int k0 = 0; k0 < K; k0 += 32, wk += (size_t)32 * N) {
    // ---- issue all loads for this k-step (distinct regs, clause-able) ----
    float wr[16];
#pragma unroll
    for (int j = 0; j < 16; ++j) wr[j] = wbase[wk + (size_t)(j + (j & 8)) * N];
    v8h l0 = *(const v8h*)(Ah + a0 + k0), h0 = *(const v8h*)(Ah + a0 + k0 + 16);
    v8h l1 = *(const v8h*)(Ah + a1 + k0), h1 = *(const v8h*)(Ah + a1 + k0 + 16);
    v8h l2 = *(const v8h*)(Ah + a2 + k0), h2 = *(const v8h*)(Ah + a2 + k0 + 16);
    v8h l3 = *(const v8h*)(Ah + a3 + k0), h3 = *(const v8h*)(Ah + a3 + k0 + 16);
    __builtin_prefetch(wbase + wk + (size_t)32 * N, 0, 1);  // speculative
    // ---- convert B fragment ----
    v16h bf;
#pragma unroll
    for (int j = 0; j < 16; ++j) bf[j] = (_Float16)wr[j];
    // ---- 4 WMMAs ----
    acc0 = wmma16x16x32(cat16(l0, h0), bf, acc0);
    acc1 = wmma16x16x32(cat16(l1, h1), bf, acc1);
    acc2 = wmma16x16x32(cat16(l2, h2), bf, acc2);
    acc3 = wmma16x16x32(cat16(l3, h3), bf, acc3);
  }
  const float bb = bias ? bias[n0 + l15] : 0.0f;
#define ST_TILE(T, A)                                                        \
  do {                                                                       \
    _Pragma("unroll") for (int r = 0; r < 8; ++r)                            \
        C[(size_t)(((T) << 4) + r + (half << 3)) * N + n0 + l15] =           \
            (A)[r] + bb;                                                     \
  } while (0)
  ST_TILE(0, acc0);
  ST_TILE(1, acc1);
  ST_TILE(2, acc2);
  ST_TILE(3, acc3);
#undef ST_TILE
}

// ---------------------------------------------------------------------------
// RoPE on [64, nheads*128]; thread owns the (d, d+64) pair.
// If dst_h != nullptr, results are written there as f16; else f32 in-place.
// ---------------------------------------------------------------------------
__global__ void rope_kernel(const float* __restrict__ Xin,
                            float* __restrict__ Xout,
                            _Float16* __restrict__ dst_h,
                            const int* __restrict__ pos_ids,
                            const int nheads) {
  const int idx   = blockIdx.x * blockDim.x + threadIdx.x;
  const int total = M_ * nheads * 64;
  if (idx >= total) return;
  const int d   = idx & 63;
  const int h   = (idx >> 6) % nheads;
  const int row = idx / (nheads << 6);
  const float pos = (float)pos_ids[row];            // rows are exactly [b*16+s]
  // inv_freq = 1e6^(-d/64) = exp(-(d/64)*ln(1e6))
  const float ang = pos * expf(-13.815510557964274f * (float)d * (1.0f / 64.0f));
  float sn, cs;
  sincosf(ang, &sn, &cs);
  const size_t off = (size_t)row * ((size_t)nheads << 7) + (h << 7) + d;
  const float x1 = Xin[off], x2 = Xin[off + 64];
  const float o1 = x1 * cs - x2 * sn;
  const float o2 = x2 * cs + x1 * sn;
  if (dst_h) {
    dst_h[off]      = (_Float16)o1;
    dst_h[off + 64] = (_Float16)o2;
  } else {
    Xout[off]      = o1;
    Xout[off + 64] = o2;
  }
}

// ---------------------------------------------------------------------------
// Scores: per (b,h,32-key chunk): S[16,32] = q[16,128] * k^T, scale + mask.
// q is pre-staged f16; keys come from the fp32 cache or the new-key buffer.
// Fully unrolled d-loop -> all loads are immediate offsets off 3 bases.
// ---------------------------------------------------------------------------
__global__ void attn_scores(const _Float16* __restrict__ qh,  // [64][4096] f16
                            const float* __restrict__ kbuf,   // [64][1024]
                            const float* __restrict__ cache_k,// [B][HKV][PAST][128]
                            float* __restrict__ scores) {     // [B][H][16][4096]
  const int chunk = blockIdx.x;                // 32 keys
  const int h = blockIdx.y, b = blockIdx.z;
  const int hkv = h >> 2;
  const int lane = threadIdx.x, half = lane >> 4, l15 = lane & 15;

  const int key0 = (chunk << 5) + l15;
  const int key1 = key0 + 16;
  const float* krow0 = ((key0 < PAST_)
      ? cache_k + ((size_t)(b * HKV_ + hkv) * PAST_ + key0) * D_
      : kbuf + (size_t)((b << 4) + (key0 - PAST_)) * (HKV_ * D_) + (hkv << 7))
      + (half << 4);
  const float* krow1 = ((key1 < PAST_)
      ? cache_k + ((size_t)(b * HKV_ + hkv) * PAST_ + key1) * D_
      : kbuf + (size_t)((b << 4) + (key1 - PAST_)) * (HKV_ * D_) + (hkv << 7))
      + (half << 4);
  const _Float16* qrow =
      qh + (size_t)((b << 4) + l15) * HID_ + (h << 7) + (half << 4);

  v8f acc0 = {}, acc1 = {};
#pragma unroll
  for (int d0 = 0; d0 < D_; d0 += 32) {
    v8h alo = *(const v8h*)(qrow + d0);
    v8h ahi = *(const v8h*)(qrow + d0 + 8);
    v4f y00 = ((const v4f*)(krow0 + d0))[0];
    v4f y01 = ((const v4f*)(krow0 + d0))[1];
    v4f y02 = ((const v4f*)(krow0 + d0))[2];
    v4f y03 = ((const v4f*)(krow0 + d0))[3];
    v4f y10 = ((const v4f*)(krow1 + d0))[0];
    v4f y11 = ((const v4f*)(krow1 + d0))[1];
    v4f y12 = ((const v4f*)(krow1 + d0))[2];
    v4f y13 = ((const v4f*)(krow1 + d0))[3];
    v16h bf0, bf1;
#pragma unroll
    for (int j = 0; j < 4; ++j) {
      bf0[j] = (_Float16)y00[j];      bf0[4 + j] = (_Float16)y01[j];
      bf0[8 + j] = (_Float16)y02[j];  bf0[12 + j] = (_Float16)y03[j];
      bf1[j] = (_Float16)y10[j];      bf1[4 + j] = (_Float16)y11[j];
      bf1[8 + j] = (_Float16)y12[j];  bf1[12 + j] = (_Float16)y13[j];
    }
    v16h af = cat16(alo, ahi);
    acc0 = wmma16x16x32(af, bf0, acc0);
    acc1 = wmma16x16x32(af, bf1, acc1);
  }
  const float scale = 0.08838834764831845f;   // 1/sqrt(128)
  float* srow = scores + (size_t)((b * H_ + h) << 4) * KLEN_;
#pragma unroll
  for (int r = 0; r < 8; ++r) {
    const int qi = r + (half << 3);
    const float v0 = (key0 <= PAST_ + qi) ? acc0[r] * scale : -1.0e30f;
    const float v1 = (key1 <= PAST_ + qi) ? acc1[r] * scale : -1.0e30f;
    srow[(size_t)qi * KLEN_ + key0] = v0;
    srow[(size_t)qi * KLEN_ + key1] = v1;
  }
}

// ---------------------------------------------------------------------------
// Row softmax over 4096; emits f16 probabilities (A-fragment friendly layout).
// ---------------------------------------------------------------------------
__global__ void softmax_kernel(const float* __restrict__ scores,
                               _Float16* __restrict__ P) {
  const int row = blockIdx.x;                 // 0..2047
  const float* src = scores + (size_t)row * KLEN_;
  _Float16* dst = P + (size_t)row * KLEN_;
  const int tid = threadIdx.x;
  const int lane = tid & 31, wid = tid >> 5;

  float v[16];
  float lmax = -1.0e30f;
#pragma unroll
  for (int i = 0; i < 16; ++i) {
    v[i] = src[tid + (i << 8)];
    lmax = fmaxf(lmax, v[i]);
  }
  __shared__ float smax[8];
  __shared__ float ssum[8];
#pragma unroll
  for (int off = 16; off > 0; off >>= 1)
    lmax = fmaxf(lmax, __shfl_xor(lmax, off, 32));
  if (lane == 0) smax[wid] = lmax;
  __syncthreads();
  float rowmax = smax[0];
#pragma unroll
  for (int i = 1; i < 8; ++i) rowmax = fmaxf(rowmax, smax[i]);

  float lsum = 0.0f;
#pragma unroll
  for (int i = 0; i < 16; ++i) {
    v[i] = expf(v[i] - rowmax);
    lsum += v[i];
  }
#pragma unroll
  for (int off = 16; off > 0; off >>= 1)
    lsum += __shfl_xor(lsum, off, 32);
  if (lane == 0) ssum[wid] = lsum;
  __syncthreads();
  float rowsum = 0.0f;
#pragma unroll
  for (int i = 0; i < 8; ++i) rowsum += ssum[i];
  const float inv = 1.0f / rowsum;
#pragma unroll
  for (int i = 0; i < 16; ++i)
    dst[tid + (i << 8)] = (_Float16)(v[i] * inv);
}

// ---------------------------------------------------------------------------
// AV: per (b,h,16-col tile of D): O[16,16] = P[16,4096] * V[4096,16-slice].
// Main loop covers keys [0,4064) (pure cache, base+offset addressing);
// the tail step (keys 4064..4095) resolves the cache/new split per lane-half.
// Output stored as f16 for the O-projection's A side.
// ---------------------------------------------------------------------------
__global__ void attn_av(const _Float16* __restrict__ P,      // [B][H][16][4096]
                        const float* __restrict__ vbuf,      // [64][1024]
                        const float* __restrict__ cache_v,   // [B][HKV][PAST][128]
                        _Float16* __restrict__ obuf_h) {     // [64][4096] f16
  const int nt = blockIdx.x;                  // d-tile 0..7
  const int h = blockIdx.y, b = blockIdx.z;
  const int hkv = h >> 2;
  const int lane = threadIdx.x, half = lane >> 4, l15 = lane & 15;

  const size_t poff =
      (size_t)(((b * H_ + h) << 4) + l15) * KLEN_ + (half << 4);
  const float* vbase = cache_v +
      ((size_t)(b * HKV_ + hkv) * PAST_ + (half << 4)) * D_ + (nt << 4) + l15;

  v8f acc = {};
  size_t vk = 0;
  for (int k0 = 0; k0 < PAST_ - 16; k0 += 32, vk += (size_t)32 * D_) {
    float vr[16];
#pragma unroll
    for (int j = 0; j < 16; ++j) vr[j] = vbase[vk + (size_t)j * D_];
    v8h a0 = *(const v8h*)(P + poff + k0);
    v8h a1 = *(const v8h*)(P + poff + k0 + 8);
    v16h bf;
#pragma unroll
    for (int j = 0; j < 16; ++j) bf[j] = (_Float16)vr[j];
    acc = wmma16x16x32(cat16(a0, a1), bf, acc);
  }
  {   // tail: k0 = 4064; half 0 -> cache keys 4064..4079, half 1 -> new keys
    const int k0 = PAST_ - 16;
    const float* bbase;
    size_t bstride;
    if (half == 0) {
      bbase = cache_v + ((size_t)(b * HKV_ + hkv) * PAST_ + k0) * D_ +
              (nt << 4) + l15;
      bstride = D_;
    } else {
      bbase = vbuf + (size_t)(b << 4) * (HKV_ * D_) + (hkv << 7) +
              (nt << 4) + l15;
      bstride = HKV_ * D_;
    }
    float vr[16];
#pragma unroll
    for (int j = 0; j < 16; ++j) vr[j] = bbase[(size_t)j * bstride];
    v8h a0 = *(const v8h*)(P + poff + k0);
    v8h a1 = *(const v8h*)(P + poff + k0 + 8);
    v16h bf;
#pragma unroll
    for (int j = 0; j < 16; ++j) bf[j] = (_Float16)vr[j];
    acc = wmma16x16x32(cat16(a0, a1), bf, acc);
  }
#pragma unroll
  for (int r = 0; r < 8; ++r) {
    const int qi = r + (half << 3);
    obuf_h[(size_t)((b << 4) + qi) * HID_ + (h << 7) + (nt << 4) + l15] =
        (_Float16)acc[r];
  }
}

// ---------------------------------------------------------------------------
extern "C" void kernel_launch(void* const* d_in, const int* in_sizes, int n_in,
                              void* d_out, int out_size, void* d_ws, size_t ws_size,
                              hipStream_t stream) {
  const float* hidden  = (const float*)d_in[0];
  const int*   pos     = (const int*)d_in[1];
  const float* cache_k = (const float*)d_in[2];
  const float* cache_v = (const float*)d_in[3];
  const float* Wq = (const float*)d_in[4];
  const float* bq = (const float*)d_in[5];
  const float* Wk = (const float*)d_in[6];
  const float* bk = (const float*)d_in[7];
  const float* Wv = (const float*)d_in[8];
  const float* bv = (const float*)d_in[9];
  const float* Wo = (const float*)d_in[10];
  float* out = (float*)d_out;

  // Workspace carve-up (~51 MB total)
  float* ws     = (float*)d_ws;
  float* qbuf   = ws;                              // 64*4096 f32
  float* kbuf   = qbuf + (size_t)M_ * HID_;        // 64*1024 f32
  float* vbuf   = kbuf + (size_t)M_ * 1024;        // 64*1024 f32
  float* scores = vbuf + (size_t)M_ * 1024;        // 2048*4096 f32
  _Float16* P   = (_Float16*)(scores + (size_t)(B_ * H_ * S_) * KLEN_);
  _Float16* xh  = P + (size_t)(B_ * H_ * S_) * KLEN_;  // 64*4096 f16
  _Float16* qh  = xh + (size_t)M_ * HID_;              // 64*4096 f16
  _Float16* oh  = qh + (size_t)M_ * HID_;              // 64*4096 f16

  // 0) stage hidden as f16
  f32_to_f16_kernel<<<dim3((M_ * HID_ / 4 + 255) / 256), 256, 0, stream>>>(
      hidden, xh, M_ * HID_ / 4);
  // 1) QKV projections (+bias fused)
  gemm_f16wmma<<<dim3(HID_ / 16), 32, 0, stream>>>(xh, Wq, bq, qbuf, HID_, HID_);
  gemm_f16wmma<<<dim3(1024 / 16), 32, 0, stream>>>(xh, Wk, bk, kbuf, HID_, 1024);
  gemm_f16wmma<<<dim3(1024 / 16), 32, 0, stream>>>(xh, Wv, bv, vbuf, HID_, 1024);
  // 2) RoPE: q -> f16 staging buffer, k -> f32 in place
  rope_kernel<<<dim3((M_ * H_ * 64 + 255) / 256), 256, 0, stream>>>(
      qbuf, nullptr, qh, pos, H_);
  rope_kernel<<<dim3((M_ * HKV_ * 64 + 255) / 256), 256, 0, stream>>>(
      kbuf, kbuf, nullptr, pos, HKV_);
  // 3) scores + causal mask
  attn_scores<<<dim3(KLEN_ / 32, H_, B_), 32, 0, stream>>>(qh, kbuf, cache_k,
                                                           scores);
  // 4) softmax -> f16 probabilities
  softmax_kernel<<<dim3(B_ * H_ * S_), 256, 0, stream>>>(scores, P);
  // 5) P * V -> f16 output
  attn_av<<<dim3(D_ / 16, H_, B_), 32, 0, stream>>>(P, vbuf, cache_v, oh);
  // 6) output projection
  gemm_f16wmma<<<dim3(HID_ / 16), 32, 0, stream>>>(oh, Wo, nullptr, out, HID_,
                                                   HID_);
}
